// MoGCN_58265526337657
// MI455X (gfx1250) — matrix operand order repaired and checked by reading, not compiled, not run on hardware
//
#include <hip/hip_runtime.h>

// ---------------------------------------------------------------------------
// MoGCN on MI455X (gfx1250, wave32).
//  - Layer-1 GEMM: fp32 X staged->bf16 LDS, v_wmma_f32_16x16x32_bf16.
//  - Layer-2/fusion GEMMs: bf16 activations copied straight to LDS via
//    global_load_async_to_lds_b128 (ASYNCcnt), weights converted in-kernel.
//  - Edge aggregation: float4 gather + global_atomic_add_f32.
// ---------------------------------------------------------------------------

#define GCN_N 50000
#define GCN_E 600000
#define GCN_H1 128
#define GCN_H2 64

typedef __bf16        v16bf __attribute__((ext_vector_type(16)));
typedef float         v8f   __attribute__((ext_vector_type(8)));
typedef unsigned int  u32x4 __attribute__((ext_vector_type(4)));

union Frag16 { u32x4 q[2]; v16bf v; };

__device__ __forceinline__ unsigned short f2bf(float f) {
  unsigned u = __float_as_uint(f);
  unsigned rnd = 0x7FFFu + ((u >> 16) & 1u);           // round-to-nearest-even
  return (unsigned short)((u + rnd) >> 16);
}
__device__ __forceinline__ unsigned pack2bf(float a, float b) {
  return (unsigned)f2bf(a) | ((unsigned)f2bf(b) << 16);
}

// ---------------------------------------------------------------------------
// Utility kernels
// ---------------------------------------------------------------------------
__global__ void fill_kernel(float* __restrict__ p, float v, int n) {
  int i = blockIdx.x * blockDim.x + threadIdx.x;
  if (i < n) p[i] = v;
}

__global__ void deg_kernel(const int* __restrict__ ei, const float* __restrict__ ew,
                           float* __restrict__ deg, int E) {
  int e = blockIdx.x * blockDim.x + threadIdx.x;
  if (e < E) atomicAdd(&deg[ei[E + e]], ew[e]);        // dst = ei[1][e]
}

__global__ void rsqrt_kernel(float* __restrict__ d, int n) {
  int i = blockIdx.x * blockDim.x + threadIdx.x;
  if (i < n) { float v = d[i]; d[i] = (v > 0.0f) ? rsqrtf(v) : 0.0f; }
}

// ---------------------------------------------------------------------------
// GEMM common geometry: 256 threads (8 waves), 64 rows x H cols per block,
// H = 32*CPW.  Wave w -> row tile (w>>1), col group (w&1) of CPW tiles.
// ---------------------------------------------------------------------------

// ---- Variant 1: fp32 input (layer-1 features) ------------------------------
template<int CPW>
__global__ __launch_bounds__(256) void gemm_wmma_f32in_kernel(
    const float* __restrict__ X, const float* __restrict__ W,
    float* __restrict__ Y, int Nrows, int K)           // K % 4 == 0
{
  constexpr int H  = 32 * CPW;
  constexpr int BV = H / 32;
  __shared__ __align__(16) unsigned short lA[64 * 32];
  __shared__ __align__(16) unsigned short lB[H * 32];

  const int t    = threadIdx.x;
  const int wave = t >> 5;
  const int lane = t & 31;
  const int rt   = wave >> 1;
  const int cg   = wave & 1;
  const int row0 = blockIdx.x * 64;
  const int m    = lane & 15;
  const int hh   = lane >> 4;

  const int arow = t >> 2;
  const int ak   = (t & 3) * 8;
  const bool arv = (row0 + arow) < Nrows;
  const int agr  = min(row0 + arow, Nrows - 1);
  const float4 zero4 = make_float4(0.f, 0.f, 0.f, 0.f);

  float4 av[2];
  float4 bv[BV];

  auto loadA = [&](int k0) {
#pragma unroll
    for (int j = 0; j < 2; ++j) {
      int gk  = k0 + ak + j * 4;
      int gkc = min(gk, K - 4);
      float4 v = *reinterpret_cast<const float4*>(&X[agr * K + gkc]);
      av[j] = (arv && gk < K) ? v : zero4;
    }
  };
  auto storeA = [&]() {
    u32x4 pk;
    pk.x = pack2bf(av[0].x, av[0].y);
    pk.y = pack2bf(av[0].z, av[0].w);
    pk.z = pack2bf(av[1].x, av[1].y);
    pk.w = pack2bf(av[1].z, av[1].w);
    *reinterpret_cast<u32x4*>(&lA[t * 8]) = pk;
  };
  auto loadB = [&](int k0) {
#pragma unroll
    for (int i = 0; i < BV; ++i) {
      int qi = i * 256 + t;
      int n  = (qi % (H / 4)) * 4;
      int k  = qi / (H / 4);
      int gk  = k0 + k;
      int gkc = min(gk, K - 1);
      float4 v = *reinterpret_cast<const float4*>(&W[gkc * H + n]);
      bv[i] = (gk < K) ? v : zero4;
    }
  };
  auto storeB = [&]() {
#pragma unroll
    for (int i = 0; i < BV; ++i) {
      int qi = i * 256 + t;
      int n  = (qi % (H / 4)) * 4;
      int k  = qi / (H / 4);
      lB[(n + 0) * 32 + k] = f2bf(bv[i].x);
      lB[(n + 1) * 32 + k] = f2bf(bv[i].y);
      lB[(n + 2) * 32 + k] = f2bf(bv[i].z);
      lB[(n + 3) * 32 + k] = f2bf(bv[i].w);
    }
  };

  v8f acc[CPW];
#pragma unroll
  for (int c = 0; c < CPW; ++c)
#pragma unroll
    for (int j = 0; j < 8; ++j) acc[c][j] = 0.0f;

  loadA(0);
  loadB(0);

  for (int k0 = 0; k0 < K; k0 += 32) {
    __syncthreads();
    storeA();
    storeB();
    __syncthreads();
    if (k0 + 32 < K) { loadA(k0 + 32); loadB(k0 + 32); }

    Frag16 a;
    const u32x4* pa = reinterpret_cast<const u32x4*>(&lA[(rt * 16 + m) * 32 + 8 * hh]);
    a.q[0] = pa[0];
    a.q[1] = pa[2];
#pragma unroll
    for (int c = 0; c < CPW; ++c) {
      int n = (cg * CPW + c) * 16 + m;
      Frag16 b;
      const u32x4* pb = reinterpret_cast<const u32x4*>(&lB[n * 32 + 8 * hh]);
      b.q[0] = pb[0];
      b.q[1] = pb[2];
      acc[c] = __builtin_amdgcn_wmma_f32_16x16x32_bf16(
          false, a.v, false, b.v, (short)0, acc[c], false, false);
    }
  }

#pragma unroll
  for (int c = 0; c < CPW; ++c) {
    int col = (cg * CPW + c) * 16 + m;
#pragma unroll
    for (int r = 0; r < 8; ++r) {
      int grow = row0 + rt * 16 + r + 8 * hh;
      if (grow < Nrows) Y[grow * H + col] = acc[c][r];
    }
  }
}

// ---- Variant 2: bf16 input (layer-2 / fusion), async A staging -------------
template<int CPW, bool BIAS_RELU>
__global__ __launch_bounds__(256) void gemm_wmma_bf16in_kernel(
    const unsigned short* __restrict__ X16, const float* __restrict__ W,
    const float* __restrict__ bias, float* __restrict__ Y,
    int Nrows, int K)                                  // K % 32 == 0
{
  constexpr int H  = 32 * CPW;
  constexpr int BV = H / 32;
  __shared__ __align__(16) unsigned short lA[64 * 32];
  __shared__ __align__(16) unsigned short lB[H * 32];

  const int t    = threadIdx.x;
  const int wave = t >> 5;
  const int lane = t & 31;
  const int rt   = wave >> 1;
  const int cg   = wave & 1;
  const int row0 = blockIdx.x * 64;
  const int m    = lane & 15;
  const int hh   = lane >> 4;

  const int arow = t >> 2;
  const int ak   = (t & 3) * 8;
  const int agr  = min(row0 + arow, Nrows - 1);        // clamp; junk rows masked in epilogue
  const unsigned ldsA = (unsigned)(size_t)&lA[t * 8];  // LDS byte offset of this thread's 16B

  float4 bv[BV];
  auto loadB = [&](int k0) {
#pragma unroll
    for (int i = 0; i < BV; ++i) {
      int qi = i * 256 + t;
      int n  = (qi % (H / 4)) * 4;
      int k  = qi / (H / 4);
      bv[i] = *reinterpret_cast<const float4*>(&W[(k0 + k) * H + n]);
    }
  };
  auto storeB = [&]() {
#pragma unroll
    for (int i = 0; i < BV; ++i) {
      int qi = i * 256 + t;
      int n  = (qi % (H / 4)) * 4;
      int k  = qi / (H / 4);
      lB[(n + 0) * 32 + k] = f2bf(bv[i].x);
      lB[(n + 1) * 32 + k] = f2bf(bv[i].y);
      lB[(n + 2) * 32 + k] = f2bf(bv[i].z);
      lB[(n + 3) * 32 + k] = f2bf(bv[i].w);
    }
  };

  v8f acc[CPW];
#pragma unroll
  for (int c = 0; c < CPW; ++c)
#pragma unroll
    for (int j = 0; j < 8; ++j) acc[c][j] = 0.0f;

  loadB(0);

  for (int k0 = 0; k0 < K; k0 += 32) {
    __syncthreads();                                   // prev LDS reads done
    // Async copy this thread's 16B of the 64x32 bf16 A tile: global -> LDS.
    {
      unsigned voff = (unsigned)((agr * K + k0 + ak) * 2);
      asm volatile("global_load_async_to_lds_b128 %0, %1, %2"
                   :: "v"(ldsA), "v"(voff), "s"(X16) : "memory");
    }
    storeB();
    if (k0 + 32 < K) loadB(k0 + 32);                   // prefetch next W chunk
    asm volatile("s_wait_asynccnt 0x0" ::: "memory");  // A tile landed in LDS
    __syncthreads();

    Frag16 a;
    const u32x4* pa = reinterpret_cast<const u32x4*>(&lA[(rt * 16 + m) * 32 + 8 * hh]);
    a.q[0] = pa[0];
    a.q[1] = pa[2];
#pragma unroll
    for (int c = 0; c < CPW; ++c) {
      int n = (cg * CPW + c) * 16 + m;
      Frag16 b;
      const u32x4* pb = reinterpret_cast<const u32x4*>(&lB[n * 32 + 8 * hh]);
      b.q[0] = pb[0];
      b.q[1] = pb[2];
      acc[c] = __builtin_amdgcn_wmma_f32_16x16x32_bf16(
          false, a.v, false, b.v, (short)0, acc[c], false, false);
    }
  }

#pragma unroll
  for (int c = 0; c < CPW; ++c) {
    int col = (cg * CPW + c) * 16 + m;
    float bvv = BIAS_RELU ? bias[col] : 0.0f;
#pragma unroll
    for (int r = 0; r < 8; ++r) {
      int grow = row0 + rt * 16 + r + 8 * hh;
      if (grow < Nrows) {
        float v = acc[c][r];
        if (BIAS_RELU) v = fmaxf(v + bvv, 0.0f);
        Y[grow * H + col] = v;
      }
    }
  }
}

// ---------------------------------------------------------------------------
// Edge aggregation: acc[dst] += dinv[src]*ew*dinv[dst] * Y[src]  (per channel)
// ---------------------------------------------------------------------------
template<int H>
__global__ __launch_bounds__(256) void agg_kernel(
    const int* __restrict__ ei, const float* __restrict__ ew,
    const float* __restrict__ dinv, const float* __restrict__ Y,
    float* __restrict__ acc, int E)
{
  constexpr int Q = H / 4;
  int gid = blockIdx.x * blockDim.x + threadIdx.x;
  if (gid >= E * Q) return;
  int e = gid / Q, q = gid - e * Q;
  int src = ei[e], dst = ei[E + e];
  float norm = dinv[src] * ew[e] * dinv[dst];
  const float4 v = *reinterpret_cast<const float4*>(&Y[src * H + q * 4]);
  float* a = &acc[dst * H + q * 4];
  atomicAdd(a + 0, norm * v.x);
  atomicAdd(a + 1, norm * v.y);
  atomicAdd(a + 2, norm * v.z);
  atomicAdd(a + 3, norm * v.w);
}

// ---------------------------------------------------------------------------
// Self-loop + bias + ReLU -> bf16 activations:
//   out16[i, colOff+h] = bf16(relu(acc + dinv[i]^2*Y + b[h]))
// ---------------------------------------------------------------------------
__global__ void brs16_kernel(const float* __restrict__ acc, const float* __restrict__ Y,
                             const float* __restrict__ dinv, const float* __restrict__ b,
                             unsigned short* __restrict__ out16,
                             int N, int H, int outStride, int colOff)
{
  int gid = blockIdx.x * blockDim.x + threadIdx.x;
  if (gid >= N * H) return;
  int i = gid / H, h = gid - i * H;
  float s = dinv[i];
  float v = acc[gid] + s * s * Y[gid] + b[h];
  out16[i * outStride + colOff + h] = f2bf(fmaxf(v, 0.0f));
}

// ---------------------------------------------------------------------------
// Prediction head: out[i,0:2] = fused[i,:] @ W_pred + b_pred
// ---------------------------------------------------------------------------
__global__ void pred_kernel(const float* __restrict__ fused, const float* __restrict__ Wp,
                            const float* __restrict__ bp, float* __restrict__ out, int N)
{
  int i = blockIdx.x * blockDim.x + threadIdx.x;
  if (i >= N) return;
  float s0 = bp[0], s1 = bp[1];
  const float* f = &fused[i * GCN_H2];
#pragma unroll
  for (int k = 0; k < GCN_H2; ++k) {
    float x = f[k];
    s0 += x * Wp[k * 2 + 0];
    s1 += x * Wp[k * 2 + 1];
  }
  out[i * 2 + 0] = s0;
  out[i * 2 + 1] = s1;
}

// ---------------------------------------------------------------------------
// Host orchestration
// ---------------------------------------------------------------------------
static inline int cdiv(int a, int b) { return (a + b - 1) / b; }

extern "C" void kernel_launch(void* const* d_in, const int* in_sizes, int n_in,
                              void* d_out, int out_size, void* d_ws, size_t ws_size,
                              hipStream_t stream)
{
  (void)in_sizes; (void)n_in; (void)out_size; (void)ws_size;
  const int N = GCN_N, E = GCN_E;
  const int dims[3] = {1000, 1000, 500};

  // Workspace: dinv[N]f | bufA[N*128]f | bufB[N*128]f | act1[N*128]bf16 | comb[N*192]bf16
  float* ws   = (float*)d_ws;
  float* dinv = ws;
  float* bufA = dinv + N;
  float* bufB = bufA + N * GCN_H1;
  unsigned short* act1   = (unsigned short*)(bufB + N * GCN_H1);
  unsigned short* comb16 = act1 + N * GCN_H1;
  float* fusd = bufA;                    // fp32 fused activations reuse bufA

  const float* W_fuse = (const float*)d_in[21];
  const float* b_fuse = (const float*)d_in[22];
  const float* W_pred = (const float*)d_in[23];
  const float* b_pred = (const float*)d_in[24];

  for (int br = 0; br < 3; ++br) {
    const float* feat = (const float*)d_in[br * 7 + 0];
    const int*   ei   = (const int*)  d_in[br * 7 + 1];
    const float* ew   = (const float*)d_in[br * 7 + 2];
    const float* W1   = (const float*)d_in[br * 7 + 3];
    const float* b1   = (const float*)d_in[br * 7 + 4];
    const float* W2   = (const float*)d_in[br * 7 + 5];
    const float* b2   = (const float*)d_in[br * 7 + 6];
    const int d = dims[br];

    // Degrees: deg = 1 (self loop) + sum_in(ew); dinv = rsqrt(deg)
    fill_kernel<<<cdiv(N, 256), 256, 0, stream>>>(dinv, 1.0f, N);
    deg_kernel<<<cdiv(E, 256), 256, 0, stream>>>(ei, ew, dinv, E);
    rsqrt_kernel<<<cdiv(N, 256), 256, 0, stream>>>(dinv, N);

    // Layer 1: Y1 = X @ W1 ; aggregate ; +b1, relu -> bf16 act1
    gemm_wmma_f32in_kernel<4><<<cdiv(N, 64), 256, 0, stream>>>(feat, W1, bufA, N, d);
    fill_kernel<<<cdiv(N * GCN_H1, 256), 256, 0, stream>>>(bufB, 0.0f, N * GCN_H1);
    agg_kernel<GCN_H1><<<cdiv(E * (GCN_H1 / 4), 256), 256, 0, stream>>>(ei, ew, dinv, bufA, bufB, E);
    brs16_kernel<<<cdiv(N * GCN_H1, 256), 256, 0, stream>>>(bufB, bufA, dinv, b1, act1,
                                                            N, GCN_H1, GCN_H1, 0);

    // Layer 2: Y2 = act1 @ W2 (bf16 A, async LDS) ; aggregate ; -> comb16[:, br*64]
    gemm_wmma_bf16in_kernel<2, false><<<cdiv(N, 64), 256, 0, stream>>>(act1, W2, nullptr,
                                                                       bufA, N, GCN_H1);
    fill_kernel<<<cdiv(N * GCN_H2, 256), 256, 0, stream>>>(bufB, 0.0f, N * GCN_H2);
    agg_kernel<GCN_H2><<<cdiv(E * (GCN_H2 / 4), 256), 256, 0, stream>>>(ei, ew, dinv, bufA, bufB, E);
    brs16_kernel<<<cdiv(N * GCN_H2, 256), 256, 0, stream>>>(bufB, bufA, dinv, b2, comb16,
                                                            N, GCN_H2, 3 * GCN_H2, br * GCN_H2);
  }

  // Fusion: fused = relu(comb @ W_fuse + b_fuse)   [N,192]bf16 x [192,64]
  gemm_wmma_bf16in_kernel<2, true><<<cdiv(N, 64), 256, 0, stream>>>(comb16, W_fuse, b_fuse,
                                                                    fusd, N, 3 * GCN_H2);

  // Prediction: out = fused @ W_pred + b_pred       [N,64] x [64,2]
  pred_kernel<<<cdiv(N, 256), 256, 0, stream>>>(fusd, W_pred, b_pred, (float*)d_out, N);
}
